// Block_67310727463585
// MI455X (gfx1250) — matrix operand order repaired and checked by reading, not compiled, not run on hardware
//
#include <hip/hip_runtime.h>
#include <hip/hip_bf16.h>

// ---- problem constants ----
#define BB 128   // batch
#define TT 512   // time
#define CC 1024  // channels

typedef __attribute__((ext_vector_type(16))) _Float16 v16h;  // 32B
typedef __attribute__((ext_vector_type(8)))  _Float16 v8h;   // 16B
typedef __attribute__((ext_vector_type(4)))  _Float16 v4h;   // 8B
typedef __attribute__((ext_vector_type(8)))  float    v8f;
typedef __attribute__((ext_vector_type(4)))  float    v4f;
typedef __attribute__((ext_vector_type(4)))  unsigned int v4u;
typedef __attribute__((ext_vector_type(8)))  int          v8i;
typedef __attribute__((ext_vector_type(4)))  int          v4i;

#define AS1 __attribute__((address_space(1)))
#define AS3 __attribute__((address_space(3)))

#if __has_builtin(__builtin_amdgcn_tensor_load_to_lds) && \
    __has_builtin(__builtin_amdgcn_s_wait_tensorcnt)
#define HAVE_TDM 1
#else
#define HAVE_TDM 0
#endif
#if __has_builtin(__builtin_amdgcn_global_load_async_to_lds_b128) && \
    __has_builtin(__builtin_amdgcn_s_wait_asynccnt)
#define HAVE_ASYNC 1
#else
#define HAVE_ASYNC 0
#endif

// D = A(16x32 f16) x B(32x16 f16) + C (f32)
__device__ __forceinline__ v8f wmma16(v16h a, v16h b, v8f c) {
    return __builtin_amdgcn_wmma_f32_16x16x32_f16(
        /*neg_a=*/false, a, /*neg_b=*/false, b,
        /*c_mod=*/(short)0, c, /*reuse_a=*/false, /*reuse_b=*/false);
}

// A fragment here (contiguous k, not AMD table order): lane L: row m = L&15,
// halves 0..15 = K kb..kb+15, kb = 16*(L>>4) (matches B).
__device__ __forceinline__ v16h load_afrag(const _Float16* p /*16B aligned*/) {
    v8h lo = *(const v8h*)(p);
    v8h hi = *(const v8h*)(p + 8);
    v16h r;
#pragma unroll
    for (int j = 0; j < 8; j++) { r[j] = lo[j]; r[8 + j] = hi[j]; }
    return r;
}
// B fragment here (contiguous k as A): lane L: col n = L&15, halves 0..15 = K 16*(L>>4)..+15
// -> one contiguous 32B run.

#if HAVE_TDM
// 2-D TDM descriptor per ISA cdna5 §8.3/8.4: copy tile_h rows x tile_w f16
// elements (row stride = stride_elems) from global to contiguous LDS.
__device__ __forceinline__ void tdm_load_tile_2d(const void* gsrc, unsigned lds_addr,
                                                 unsigned tile_w, unsigned tile_h,
                                                 unsigned stride_elems) {
    unsigned long long ga = (unsigned long long)(size_t)gsrc;
    v4u g0;
    g0[0] = 1u;                                            // count=1 (valid), load, user
    g0[1] = lds_addr;                                      // lds byte address
    g0[2] = (unsigned)ga;                                  // global_addr[31:0]
    g0[3] = (unsigned)((ga >> 32) & 0x01FFFFFFu) | (2u << 30);  // addr[56:32], type=2
    unsigned td0 = stride_elems;                           // tensor_dim0
    unsigned td1 = 1u << 20;                               // tensor_dim1 (tile never OOB)
    v8i g1;
    g1[0] = (int)(1u << 16);                               // data_size=1 -> 2 bytes
    g1[1] = (int)((td0 & 0xFFFFu) << 16);                  // [63:48]=tensor_dim0 lo16
    g1[2] = (int)((td0 >> 16) | ((td1 & 0xFFFFu) << 16));  // dim0 hi | dim1 lo
    g1[3] = (int)((td1 >> 16) | (tile_w << 16));           // dim1 hi | tile_dim0
    g1[4] = (int)tile_h;                                   // tile_dim1 | tile_dim2=0
    g1[5] = (int)stride_elems;                             // tensor_dim0_stride[31:0]
    g1[6] = 0;                                             // stride hi | dim1_stride lo
    g1[7] = 0;
    v4i gz  = {0, 0, 0, 0};                                // 2-D: groups 2/3 unused
    v8i gz8 = {0, 0, 0, 0, 0, 0, 0, 0};                    // extra operand (clang-23 form)
    __builtin_amdgcn_tensor_load_to_lds(g0, g1, gz, gz, gz8, 0);
}
#endif

// ---------------------------------------------------------------------------
// 1) weight prep: Wv -> f16 (as-is), Wp -> f16 transposed (wpT[c][e] = Wp[e][c])
__global__ void k_prep_w(const float* __restrict__ Wp, const float* __restrict__ Wv,
                         _Float16* __restrict__ wv_h, _Float16* __restrict__ wpT_h) {
    int idx = blockIdx.x * 256 + threadIdx.x;        // over CC*CC
    int r = idx >> 10, col = idx & 1023;
    wv_h[idx] = (_Float16)Wv[idx];
    wpT_h[(size_t)col * CC + r] = (_Float16)Wp[idx];
}

// ---------------------------------------------------------------------------
// 2) fuse weights: M[d][c] = sum_e Wv[d][e] * Wp[e][c]   (one 16x16 tile / wave)
__global__ void __launch_bounds__(256) k_fuse_w(const _Float16* __restrict__ wv,
                                                const _Float16* __restrict__ wpT,
                                                _Float16* __restrict__ M) {
    int wave = threadIdx.x >> 5, lane = threadIdx.x & 31;
    int tile = blockIdx.x * 8 + wave;                // 4096 tiles
    int d0 = (tile >> 6) * 16, c0 = (tile & 63) * 16;
    int lm = lane & 15, kb8 = (lane >> 4) * 8, kb16 = (lane >> 4) * 16;
    v8f acc = {};
    for (int k0 = 0; k0 < CC; k0 += 32) {
        v16h a = load_afrag(wv + (size_t)(d0 + lm) * CC + k0 + kb16);
        v16h b = *(const v16h*)(wpT + (size_t)(c0 + lm) * CC + k0 + kb16);
        acc = wmma16(a, b, acc);
    }
    int dr = d0 + 8 * (lane >> 4), cc = c0 + lm;     // D: m = i+8*(lane>>4), n = lane&15
#pragma unroll
    for (int i = 0; i < 8; i++) M[(size_t)(dr + i) * CC + cc] = (_Float16)acc[i];
}

// ---------------------------------------------------------------------------
// 3) x f32 -> f16
__global__ void k_cvt_x(const float* __restrict__ x, _Float16* __restrict__ xh) {
    size_t i = ((size_t)blockIdx.x * 256 + threadIdx.x) * 4;
    v4f f = *(const v4f*)(x + i);
    v4h h = {(_Float16)f[0], (_Float16)f[1], (_Float16)f[2], (_Float16)f[3]};
    *(v4h*)(xh + i) = h;
}

// ---------------------------------------------------------------------------
// 4) big GEMM: v[r][d] = sum_c xh[r][c] * M[d][c],  r = b*TT+t (65536 rows)
//    Block = 8 waves x (64x64 tile), all waves share one N=64 B-panel that is
//    streamed through LDS in 4 K-quarters (32KB each), double-buffered via the
//    Tensor Data Mover (wave 0 issues tensor_load_to_lds; s_wait_tensorcnt +
//    barrier rotates buffers). A operand streams straight from global.
__device__ __forceinline__ void load_bpanel(const _Float16* __restrict__ Mh,
                                            _Float16* buf, int n0, int kq, int wave) {
#if HAVE_TDM
    if (wave == 0)
        tdm_load_tile_2d(Mh + (size_t)n0 * CC + kq * 256,
                         (unsigned)(size_t)(AS3 void*)buf, 256u, 64u, (unsigned)CC);
#else
    for (int idx = threadIdx.x; idx < 2048; idx += 256) {   // 64 rows x 32 chunks
        int row = idx >> 5, ch = idx & 31;
        *(v8h*)(buf + row * 256 + ch * 8) =
            *(const v8h*)(Mh + (size_t)(n0 + row) * CC + kq * 256 + ch * 8);
    }
#endif
}
__device__ __forceinline__ void wait_bpanel(int wave) {
#if HAVE_TDM
    if (wave == 0) __builtin_amdgcn_s_wait_tensorcnt(0);
#endif
    __syncthreads();
}

__global__ void __launch_bounds__(256) k_gemm_v(const _Float16* __restrict__ xh,
                                                const _Float16* __restrict__ Mh,
                                                _Float16* __restrict__ vh) {
    __shared__ __align__(16) _Float16 bpan[2][64 * 256];    // 2 x 32KB
    int wave = threadIdx.x >> 5, lane = threadIdx.x & 31;
    int m0 = (blockIdx.x >> 4) * 512 + wave * 64;           // 128 m-blocks
    int n0 = (blockIdx.x & 15) * 64;                        // 16 n-blocks
    int lm = lane & 15, kb8 = (lane >> 4) * 8, kb16 = (lane >> 4) * 16;

    load_bpanel(Mh, &bpan[0][0], n0, 0, wave);
    wait_bpanel(wave);

    v8f acc[4][4] = {};
    for (int kq = 0; kq < 4; kq++) {
        if (kq < 3) load_bpanel(Mh, &bpan[(kq + 1) & 1][0], n0, kq + 1, wave);
        const _Float16* bp = &bpan[kq & 1][0];
        if (kq < 3) {
#pragma unroll
            for (int i = 0; i < 4; i++)  // prefetch next-quarter A rows
                __builtin_prefetch(xh + (size_t)(m0 + 16 * i + lm) * CC +
                                       (kq + 1) * 256 + kb8, 0, 0);
        }
        for (int kk = 0; kk < 256; kk += 32) {
            int k0 = kq * 256 + kk;
            v16h a[4], b[4];
#pragma unroll
            for (int i = 0; i < 4; i++) {
                a[i] = load_afrag(xh + (size_t)(m0 + 16 * i + lm) * CC + k0 + kb16);
                b[i] = *(const v16h*)(bp + (size_t)(16 * i + lm) * 256 + kk + kb16);
            }
#pragma unroll
            for (int i = 0; i < 4; i++)
#pragma unroll
                for (int j = 0; j < 4; j++) acc[i][j] = wmma16(a[i], b[j], acc[i][j]);
        }
        if (kq < 3) wait_bpanel(wave);
    }

    int mr = 8 * (lane >> 4);
#pragma unroll
    for (int i = 0; i < 4; i++)
#pragma unroll
        for (int j = 0; j < 4; j++)
#pragma unroll
            for (int r = 0; r < 8; r++)
                vh[(size_t)(m0 + 16 * i + mr + r) * CC + n0 + 16 * j + lm] =
                    (_Float16)acc[i][j][r];
}

// ---------------------------------------------------------------------------
// 5) reversed zero-padded kernel: zx[c][u] = z[c][T-u] for u in (0,T], else 0
//    z[c][j] = line[j] ^ (2 + 100*sigmoid(pow_[c]))
__global__ void k_zx(const float* __restrict__ pow_, const float* __restrict__ line,
                     _Float16* __restrict__ zx) {
    int c = blockIdx.x, j = threadIdx.x;             // 1024 x 512
    float p = 2.0f + 100.0f / (1.0f + __expf(-pow_[c]));
    float lv = line[j];
    float val = (lv > 0.0f) ? __expf(p * __logf(lv)) : 0.0f;
    _Float16* row = zx + (size_t)c * (2 * TT);
    row[TT - j] = (_Float16)val;
    row[(j == 0) ? 0 : TT + j] = (_Float16)0.0f;
}

// ---------------------------------------------------------------------------
// 6) transpose v (b,t,c) -> vt (c,b,t)  via LDS 32x32 tiles
__global__ void __launch_bounds__(256) k_trans_v(const _Float16* __restrict__ v,
                                                 _Float16* __restrict__ vt) {
    __shared__ _Float16 tile[32][33];
    int t0 = blockIdx.x * 32, c0 = blockIdx.y * 32, b = blockIdx.z;
    int tx = threadIdx.x & 31, ty = threadIdx.x >> 5;
#pragma unroll
    for (int i = 0; i < 32; i += 8)
        tile[ty + i][tx] = v[((size_t)b * TT + t0 + ty + i) * CC + c0 + tx];
    __syncthreads();
#pragma unroll
    for (int i = 0; i < 32; i += 8)
        vt[((size_t)(c0 + ty + i) * BB + b) * TT + t0 + tx] = tile[tx][ty + i];
}

// ---------------------------------------------------------------------------
// 7) causal conv + gain + relu + output transpose.
//    Block = 8 waves = 8 channels; each wave: one 16(t) x 16(b) output tile,
//    K-loop over s in steps of 32; A is a Toeplitz fragment read from zx in LDS
//    (zeros beyond the diagonal make it causal with no masking). zx table is
//    brought into LDS with async global->LDS copies.
__global__ void __launch_bounds__(256) k_conv(const _Float16* __restrict__ vt,
                                              const _Float16* __restrict__ zx,
                                              const float* __restrict__ gain,
                                              float* __restrict__ out) {
    __shared__ __align__(16) _Float16 zl[8][2 * TT];   // 16 KB
    __shared__ __align__(16) float sy[16][16][8];      // 8 KB (t, b, c8)
    int wave = threadIdx.x >> 5, lane = threadIdx.x & 31;
    int cg = blockIdx.x;                 // channel group (8 ch)
    int t0 = blockIdx.y * 16;
    int b0 = blockIdx.z * 16;
    int c = cg * 8 + wave;

    {   // LDS fill: 8 channels x 1024 halves = 1024 x 16B chunks
        const _Float16* src = zx + (size_t)cg * 8 * 2 * TT;
        _Float16* dst = &zl[0][0];
#if HAVE_ASYNC
        for (int i = threadIdx.x; i < 1024; i += 256)
            __builtin_amdgcn_global_load_async_to_lds_b128(
                (AS1 v4i*)(src + (size_t)i * 8),
                (AS3 v4i*)(dst + (size_t)i * 8), 0, 0);
        __builtin_amdgcn_s_wait_asynccnt(0);
#else
        for (int i = threadIdx.x; i < 1024; i += 256)
            *(v8h*)(dst + (size_t)i * 8) = *(const v8h*)(src + (size_t)i * 8);
#endif
    }
    __syncthreads();

    int lm = lane & 15, kb8 = (lane >> 4) * 8, kb16 = (lane >> 4) * 16;
    int trow = t0 + lm;
    int nk = ((t0 + 15) >> 5) + 1;       // causal: only s-blocks with s0 <= t0+15
    const _Float16* vrow = vt + ((size_t)c * BB + b0 + lm) * TT + kb16;
    v8f acc = {};
    for (int ks = 0; ks < nk; ks++) {
        int s0 = ks * 32;
        int u0 = TT + s0 + kb16 - trow;  // in [1, 574] by construction
        v16h a;
#pragma unroll
        for (int j = 0; j < 8; j++) {
            a[j]     = zl[wave][u0 + j];        // A[t,k] = z[t - (s0+k)]
            a[8 + j] = zl[wave][u0 + 8 + j];
        }
        v16h b = *(const v16h*)(vrow + s0);      // B[k,n] = v[b0+n, s0+k, c]
        acc = wmma16(a, b, acc);
    }

    float g = gain[c];
#pragma unroll
    for (int i = 0; i < 8; i++) {
        float y = acc[i] * g;
        y = y > 0.0f ? y : 0.0f;
        sy[i + 8 * (lane >> 4)][lm][wave] = y;   // (t_local, b_local, c8)
    }
    __syncthreads();

    // coalesced-ish output: each thread writes 8 contiguous channels (32B)
    int tt = threadIdx.x >> 4, bb = threadIdx.x & 15;
    float* op = out + ((size_t)(b0 + bb) * TT + (t0 + tt)) * CC + (size_t)cg * 8;
    const v4f* sp = (const v4f*)&sy[tt][bb][0];
    ((v4f*)op)[0] = sp[0];
    ((v4f*)op)[1] = sp[1];
}

// ---------------------------------------------------------------------------
extern "C" void kernel_launch(void* const* d_in, const int* in_sizes, int n_in,
                              void* d_out, int out_size, void* d_ws, size_t ws_size,
                              hipStream_t stream) {
    const float* x     = (const float*)d_in[0];   // (B,T,C)
    const float* Wp    = (const float*)d_in[1];   // (C,C)
    const float* Wv    = (const float*)d_in[2];   // (C,C)
    const float* gain  = (const float*)d_in[3];   // (C)
    const float* pow_  = (const float*)d_in[4];   // (C)
    const float* line  = (const float*)d_in[5];   // (T)
    float* out = (float*)d_out;

    char* ws = (char*)d_ws;
    // layout (bytes); vt reuses the xh region once xh is dead.
    const size_t SZ_BTC_H = (size_t)BB * TT * CC * 2;        // 134217728
    _Float16* xh   = (_Float16*)(ws);                         // [0, 134M)  -> later vt
    _Float16* vh   = (_Float16*)(ws + SZ_BTC_H);              // [134M, 268M)
    _Float16* Mh   = (_Float16*)(ws + 2 * SZ_BTC_H);          // 2MB
    _Float16* wv_h = (_Float16*)(ws + 2 * SZ_BTC_H + (1u << 21));
    _Float16* wpT  = (_Float16*)(ws + 2 * SZ_BTC_H + (2u << 21));
    _Float16* zx   = (_Float16*)(ws + 2 * SZ_BTC_H + (3u << 21));
    _Float16* vt   = xh;  // alias: xh is dead after k_gemm_v

    // 1) weight conversion (+ Wp transpose)
    k_prep_w<<<(CC * CC) / 256, 256, 0, stream>>>(Wp, Wv, wv_h, wpT);
    // 2) fuse: M = Wv @ Wp   (WMMA)
    k_fuse_w<<<(64 * 64) / 8, 256, 0, stream>>>(wv_h, wpT, Mh);
    // 3) x -> f16
    k_cvt_x<<<((size_t)BB * TT * CC) / (256 * 4), 256, 0, stream>>>(x, xh);
    // 4) v = x @ M^T   (WMMA + TDM-staged B panel)
    k_gemm_v<<<(128 * 16), 256, 0, stream>>>(xh, Mh, vh);
    // 5) decay kernel table
    k_zx<<<CC, TT, 0, stream>>>(pow_, line, zx);
    // 6) v (b,t,c) -> vt (c,b,t)
    k_trans_v<<<dim3(TT / 32, CC / 32, BB), 256, 0, stream>>>(vh, vt);
    // 7) causal Toeplitz conv (WMMA) + gain + relu + output transpose
    k_conv<<<dim3(CC / 8, TT / 16, BB / 16), 256, 0, stream>>>(vt, zx, gain, out);
}
